// AttenBlocks3D_9964324126901
// MI455X (gfx1250) — matrix operand order repaired and checked by reading, not compiled, not run on hardware
//
#include <hip/hip_runtime.h>

typedef unsigned short u16;
typedef __attribute__((ext_vector_type(16))) __bf16 v16bf;
typedef __attribute__((ext_vector_type(8)))  float  v8f;

union BFrag { v16bf v; uint4 q[2]; };

__device__ __forceinline__ u16 f2bf(float f) {
  unsigned int u = __float_as_uint(f);
  unsigned int r = (u + 0x7FFFu + ((u >> 16) & 1u)) >> 16;
  return (u16)r;
}
__device__ __forceinline__ v8f zero8() { v8f z; for (int i = 0; i < 8; i++) z[i] = 0.f; return z; }
__device__ __forceinline__ float gelu_exact(float x) {
  return 0.5f * x * (1.0f + erff(x * 0.7071067811865476f));
}

#define L_TOK 32768
#define C_CH  192

// ---------------------------------------------------------------------------
// Pack f32 weight [N, K, T] (out-major, tap innermost) into bf16 WMMA B-fragment
// order: dst[(((t*KT+kt)*NT+nt)*32+lane)*16+h], with B layout n=lane%16,
// k_local = (h%8) + (lane/16)*8 + (h/8)*16.
// ---------------------------------------------------------------------------
__global__ void pack_b(const float* __restrict__ src, u16* __restrict__ dst,
                       int K, int Nn, int T, long total) {
  long id = (long)blockIdx.x * 256 + threadIdx.x;
  if (id >= total) return;
  int h = id & 15; long r = id >> 4;
  int lane = r & 31; r >>= 5;
  int NT = Nn >> 4;
  int nt = (int)(r % NT); r /= NT;
  int KT = K >> 5;
  int kt = (int)(r % KT);
  int t  = (int)(r / KT);
  int n = nt * 16 + (lane & 15);
  int k = kt * 32 + (h & 7) + ((lane >> 4) << 3) + ((h >> 3) << 4);
  dst[id] = f2bf(src[((long)n * K + k) * T + t]);
}

// ---------------------------------------------------------------------------
// LayerNorm1: wave per token, emit bf16.
// ---------------------------------------------------------------------------
__global__ void ln_k(const float* __restrict__ x, const float* __restrict__ g,
                     const float* __restrict__ b, u16* __restrict__ out) {
  int tok  = (blockIdx.x * blockDim.x + threadIdx.x) >> 5;
  int lane = threadIdx.x & 31;
  const float* row = x + (long)tok * C_CH;
  float v[6]; float s = 0.f;
#pragma unroll
  for (int j = 0; j < 6; j++) { v[j] = row[lane + 32 * j]; s += v[j]; }
#pragma unroll
  for (int m = 16; m >= 1; m >>= 1) s += __shfl_xor(s, m, 32);
  float mean = s * (1.0f / 192.0f);
  float var = 0.f;
#pragma unroll
  for (int j = 0; j < 6; j++) { float d = v[j] - mean; var += d * d; }
#pragma unroll
  for (int m = 16; m >= 1; m >>= 1) var += __shfl_xor(var, m, 32);
  float rstd = rsqrtf(var * (1.0f / 192.0f) + 1e-5f);
#pragma unroll
  for (int j = 0; j < 6; j++) {
    int c = lane + 32 * j;
    out[(long)tok * C_CH + c] = f2bf((v[j] - mean) * rstd * g[c] + b[c]);
  }
}

// ---------------------------------------------------------------------------
// bf16 WMMA GEMM: one wave = 16x64 strip (4 accumulators share one A frag).
// K loop fully unrolled via template KT. Epilogues:
//  0: QKV (scale q cols, bf16)  1: f32 (proj)  2: GELU->bf16 (fc1)
//  3: resid + val -> f32 (fc2 final)
// ---------------------------------------------------------------------------
template<int EPI, int KT>
__global__ void __launch_bounds__(256) gemm_k(const u16* __restrict__ A,
                                              const u16* __restrict__ Bpk,
                                              const float* __restrict__ bias,
                                              void* __restrict__ outv,
                                              const float* __restrict__ resid,
                                              int NT4, int ldA, int ldOut) {
  int lane = threadIdx.x & 31;
  int gw   = (blockIdx.x * blockDim.x + threadIdx.x) >> 5;
  int mt = gw / NT4, n4 = gw % NT4;
  int NT = NT4 * 4;
  int arow = mt * 16 + (lane & 15);
  const u16* Abase = A + (long)arow * ldA + ((lane >> 4) << 3);
  v8f acc[4];
#pragma unroll
  for (int j = 0; j < 4; j++) acc[j] = zero8();
#pragma unroll
  for (int kt = 0; kt < KT; kt++) {
    BFrag a;
    const u16* ap = Abase + kt * 32;
    a.q[0] = *(const uint4*)ap;
    a.q[1] = *(const uint4*)(ap + 16);
#pragma unroll
    for (int j = 0; j < 4; j++) {
      int nt = n4 * 4 + j;
      BFrag b;
      const u16* bp = Bpk + (((long)kt * NT + nt) * 32 + lane) * 16;
      b.q[0] = *(const uint4*)bp;
      b.q[1] = *(const uint4*)(bp + 8);
      acc[j] = __builtin_amdgcn_wmma_f32_16x16x32_bf16(false, a.v, false, b.v,
                                                       (short)0, acc[j], false, false);
    }
  }
#pragma unroll
  for (int j = 0; j < 4; j++) {
#pragma unroll
    for (int v = 0; v < 8; v++) {
      int m = mt * 16 + v + ((lane >> 4) << 3);
      int n = (n4 * 4 + j) * 16 + (lane & 15);
      float val = acc[j][v] + bias[n];
      long idx = (long)m * ldOut + n;
      if constexpr (EPI == 0) {
        if (n < 192) val *= 0.17677669529663687f;  // 1/sqrt(32) on q
        ((u16*)outv)[idx] = f2bf(val);
      } else if constexpr (EPI == 1) {
        ((float*)outv)[idx] = val;
      } else if constexpr (EPI == 2) {
        ((u16*)outv)[idx] = f2bf(gelu_exact(val));
      } else {
        ((float*)outv)[idx] = val + resid[idx];
      }
    }
  }
}

// ---------------------------------------------------------------------------
// 3x3x3 conv as implicit GEMM over 27 taps, 16x64 strip per wave.
// EPI 0: GELU->bf16, 1: f32.
// ---------------------------------------------------------------------------
template<int EPI, int KT>
__global__ void __launch_bounds__(256) conv_k(const u16* __restrict__ A,
                                              const u16* __restrict__ Wpk,
                                              const float* __restrict__ bias,
                                              void* __restrict__ outv,
                                              int NT4, int Kin, int ldOut) {
  int lane = threadIdx.x & 31;
  int gw   = (blockIdx.x * blockDim.x + threadIdx.x) >> 5;
  int mt = gw / NT4, n4 = gw % NT4;
  int NT = NT4 * 4;
  int m0 = mt * 16;
  int w0 = m0 & 63, hh = (m0 >> 6) & 63, dd = m0 >> 12;
  int wl = w0 + (lane & 15);
  v8f acc[4];
#pragma unroll
  for (int j = 0; j < 4; j++) acc[j] = zero8();
  for (int t = 0; t < 27; t++) {
    int dz = t / 9 - 1, dy = (t / 3) % 3 - 1, dx = t % 3 - 1;
    int dn = dd + dz, hn = hh + dy, wn = wl + dx;
    bool ok = ((unsigned)dn < 8u) && ((unsigned)hn < 64u) && ((unsigned)wn < 64u);
    long nrow = (long)(dn * 64 + hn) * 64 + wn;
    const u16* ab = A + nrow * Kin + ((lane >> 4) << 3);
#pragma unroll
    for (int kt = 0; kt < KT; kt++) {
      BFrag a;
      if (ok) {
        a.q[0] = *(const uint4*)(ab + kt * 32);
        a.q[1] = *(const uint4*)(ab + kt * 32 + 16);
      } else {
        a.q[0] = make_uint4(0, 0, 0, 0);
        a.q[1] = make_uint4(0, 0, 0, 0);
      }
#pragma unroll
      for (int j = 0; j < 4; j++) {
        int nt = n4 * 4 + j;
        BFrag b;
        const u16* bp = Wpk + (((long)(t * KT + kt) * NT + nt) * 32 + lane) * 16;
        b.q[0] = *(const uint4*)bp;
        b.q[1] = *(const uint4*)(bp + 8);
        acc[j] = __builtin_amdgcn_wmma_f32_16x16x32_bf16(false, a.v, false, b.v,
                                                         (short)0, acc[j], false, false);
      }
    }
  }
#pragma unroll
  for (int j = 0; j < 4; j++) {
#pragma unroll
    for (int v = 0; v < 8; v++) {
      int m = m0 + v + ((lane >> 4) << 3);
      int n = (n4 * 4 + j) * 16 + (lane & 15);
      float val = acc[j][v] + bias[n];
      long idx = (long)m * ldOut + n;
      if constexpr (EPI == 0) ((u16*)outv)[idx] = f2bf(gelu_exact(val));
      else                    ((float*)outv)[idx] = val;
    }
  }
}

// ---------------------------------------------------------------------------
// Transpose V into [win][head][d][key] so attention B-fragments are contiguous.
// ---------------------------------------------------------------------------
__global__ void vprep_k(const u16* __restrict__ qkv, u16* __restrict__ vt) {
  long id = (long)blockIdx.x * 256 + threadIdx.x;   // 64*6*32*512 total
  int key = id & 511;
  int d   = (int)((id >> 9) & 31);
  long wh = id >> 14;
  int head = (int)(wh % 6);
  int wi   = (int)(wh / 6);
  int whh = (wi >> 3) & 7, www = wi & 7;
  int dz = key >> 6, hy = (key >> 3) & 7, wx = key & 7;
  int tok = (dz * 64 + whh * 8 + hy) * 64 + www * 8 + wx;
  vt[id] = qkv[(long)tok * 576 + 384 + head * 32 + d];
}

// Relative position bias gather: bb[h][n][m] = table[rpi[n,m]*6 + h]
__global__ void bprep_k(const float* __restrict__ tab, const int* __restrict__ rpi,
                        float* __restrict__ bb) {
  long id = (long)blockIdx.x * 256 + threadIdx.x;   // 6*512*512
  int m = id & 511;
  int n = (int)((id >> 9) & 511);
  int h = (int)(id >> 18);
  bb[id] = tab[rpi[n * 512 + m] * 6 + h];
}

// ---------------------------------------------------------------------------
// Flash-style window attention: block = (window, head, 16-row*8wave rowblock).
// Wave keeps Q frag + O accumulator, streams 32-key tiles. All small loops
// fully unrolled so accumulator arrays stay statically indexed (no v_movrels).
// ---------------------------------------------------------------------------
__global__ void __launch_bounds__(256) attn_k(const u16* __restrict__ qkv,
                                              const u16* __restrict__ vt,
                                              const float* __restrict__ biasb,
                                              u16* __restrict__ attn_o) {
  __shared__ __align__(16) u16 pst[8][16 * 32];
  int lane = threadIdx.x & 31;
  int wv   = threadIdx.x >> 5;
  int wi   = blockIdx.x / 24;
  int head = (blockIdx.x / 4) % 6;
  int rb   = blockIdx.x & 3;
  int whh = (wi >> 3) & 7, www = wi & 7;
  auto gtok = [&](int n) {
    int dz = n >> 6, hy = (n >> 3) & 7, wx = n & 7;
    return (dz * 64 + whh * 8 + hy) * 64 + www * 8 + wx;
  };
  int n0 = rb * 128 + wv * 16;

  BFrag qf;
  {
    int tq = gtok(n0 + (lane & 15));
    const u16* qp = qkv + (long)tq * 576 + head * 32 + ((lane >> 4) << 3);
    qf.q[0] = *(const uint4*)qp;
    qf.q[1] = *(const uint4*)(qp + 16);
  }
  float mrun[8], lrun[8];
  v8f o0 = zero8(), o1 = zero8();
#pragma unroll
  for (int v = 0; v < 8; v++) { mrun[v] = -1e30f; lrun[v] = 0.f; }
  const float* bb = biasb + (long)head * 512 * 512;
  const u16* vbase = vt + (long)(wi * 6 + head) * 32 * 512;

  for (int kt2 = 0; kt2 < 16; kt2++) {
    v8f s[2];
#pragma unroll
    for (int ss = 0; ss < 2; ss++) {
      int key0 = kt2 * 32 + ss * 16;
      BFrag kf;
      int tk = gtok(key0 + (lane & 15));
      const u16* kp = qkv + (long)tk * 576 + 192 + head * 32 + ((lane >> 4) << 3);
      kf.q[0] = *(const uint4*)kp;
      kf.q[1] = *(const uint4*)(kp + 16);
      v8f z = zero8();
      s[ss] = __builtin_amdgcn_wmma_f32_16x16x32_bf16(false, qf.v, false, kf.v,
                                                      (short)0, z, false, false);
#pragma unroll
      for (int v = 0; v < 8; v++) {
        int rn = n0 + v + ((lane >> 4) << 3);
        int cm = key0 + (lane & 15);
        s[ss][v] += bb[(long)rn * 512 + cm];
      }
    }
    // online softmax over this 32-key slab (row reductions within 16-lane halves)
#pragma unroll
    for (int v = 0; v < 8; v++) {
      float tm = fmaxf(s[0][v], s[1][v]);
#pragma unroll
      for (int m = 8; m >= 1; m >>= 1) tm = fmaxf(tm, __shfl_xor(tm, m, 32));
      float mn = fmaxf(mrun[v], tm);
      float sc = __expf(mrun[v] - mn);
      float p0 = __expf(s[0][v] - mn);
      float p1 = __expf(s[1][v] - mn);
      float rs = p0 + p1;
#pragma unroll
      for (int m = 8; m >= 1; m >>= 1) rs += __shfl_xor(rs, m, 32);
      lrun[v] = lrun[v] * sc + rs;
      mrun[v] = mn;
      o0[v] *= sc; o1[v] *= sc;
      int r = v + ((lane >> 4) << 3);
      pst[wv][r * 32 + (lane & 15)]      = f2bf(p0);
      pst[wv][r * 32 + 16 + (lane & 15)] = f2bf(p1);
    }
    asm volatile("s_wait_dscnt 0" ::: "memory");
    BFrag pa;
    {
      int base = (lane & 15) * 32 + ((lane >> 4) << 3);
      pa.q[0] = *(const uint4*)&pst[wv][base];
      pa.q[1] = *(const uint4*)&pst[wv][base + 16];
    }
#pragma unroll
    for (int s2 = 0; s2 < 2; s2++) {
      BFrag vf;
      int d = s2 * 16 + (lane & 15);
      const u16* vp = vbase + (long)d * 512 + kt2 * 32 + ((lane >> 4) << 3);
      vf.q[0] = *(const uint4*)vp;
      vf.q[1] = *(const uint4*)(vp + 16);
      if (s2 == 0)
        o0 = __builtin_amdgcn_wmma_f32_16x16x32_bf16(false, pa.v, false, vf.v,
                                                     (short)0, o0, false, false);
      else
        o1 = __builtin_amdgcn_wmma_f32_16x16x32_bf16(false, pa.v, false, vf.v,
                                                     (short)0, o1, false, false);
    }
  }
#pragma unroll
  for (int v = 0; v < 8; v++) {
    int rn = n0 + v + ((lane >> 4) << 3);
    long tok = gtok(rn);
    float inv = 1.0f / lrun[v];
    attn_o[tok * C_CH + head * 32 + (lane & 15)]      = f2bf(o0[v] * inv);
    attn_o[tok * C_CH + head * 32 + 16 + (lane & 15)] = f2bf(o1[v] * inv);
  }
}

// Global average pool over tokens per channel.
__global__ void pool_k(const float* __restrict__ h2, float* __restrict__ pooled) {
  int c = blockIdx.x;
  float s = 0.f;
  for (int t = threadIdx.x; t < L_TOK; t += 256) s += h2[(long)t * C_CH + c];
  __shared__ float red[256];
  red[threadIdx.x] = s; __syncthreads();
  for (int st = 128; st > 0; st >>= 1) {
    if (threadIdx.x < st) red[threadIdx.x] += red[threadIdx.x + st];
    __syncthreads();
  }
  if (threadIdx.x == 0) pooled[c] = red[0] * (1.0f / (float)L_TOK);
}

// Channel attention MLP (192 -> 6 -> 192, relu/sigmoid).
__global__ void ca_k(const float* __restrict__ pooled,
                     const float* __restrict__ w1, const float* __restrict__ b1,
                     const float* __restrict__ w2, const float* __restrict__ b2,
                     float* __restrict__ a) {
  __shared__ float hid[6];
  int t = threadIdx.x;
  if (t < 6) {
    float s = b1[t];
    for (int c = 0; c < 192; c++) s += pooled[c] * w1[t * 192 + c];
    hid[t] = fmaxf(s, 0.f);
  }
  __syncthreads();
  if (t < 192) {
    float s = b2[t];
#pragma unroll
    for (int j = 0; j < 6; j++) s += hid[j] * w2[t * 6 + j];
    a[t] = 1.0f / (1.0f + __expf(-s));
  }
}

// x2 = x + proj(attn) + 0.01 * h2 * a[c]; then LayerNorm2 -> bf16.
__global__ void fuse_k(const float* __restrict__ x, float* __restrict__ x2,
                       const float* __restrict__ h2, const float* __restrict__ av,
                       const float* __restrict__ g, const float* __restrict__ b,
                       u16* __restrict__ xn2) {
  int tok  = (blockIdx.x * blockDim.x + threadIdx.x) >> 5;
  int lane = threadIdx.x & 31;
  float v[6]; float s = 0.f;
#pragma unroll
  for (int j = 0; j < 6; j++) {
    int c = lane + 32 * j;
    long idx = (long)tok * C_CH + c;
    float val = x[idx] + x2[idx] + 0.01f * h2[idx] * av[c];
    v[j] = val; x2[idx] = val; s += val;
  }
#pragma unroll
  for (int m = 16; m >= 1; m >>= 1) s += __shfl_xor(s, m, 32);
  float mean = s * (1.0f / 192.0f);
  float var = 0.f;
#pragma unroll
  for (int j = 0; j < 6; j++) { float d = v[j] - mean; var += d * d; }
#pragma unroll
  for (int m = 16; m >= 1; m >>= 1) var += __shfl_xor(var, m, 32);
  float rstd = rsqrtf(var * (1.0f / 192.0f) + 1e-5f);
#pragma unroll
  for (int j = 0; j < 6; j++) {
    int c = lane + 32 * j;
    xn2[(long)tok * C_CH + c] = f2bf((v[j] - mean) * rstd * g[c] + b[c]);
  }
}

// ---------------------------------------------------------------------------
extern "C" void kernel_launch(void* const* d_in, const int* in_sizes, int n_in,
                              void* d_out, int out_size, void* d_ws, size_t ws_size,
                              hipStream_t stream) {
  (void)in_sizes; (void)n_in; (void)out_size; (void)ws_size;
  const float* x      = (const float*)d_in[0];
  const float* n1g    = (const float*)d_in[1];
  const float* n1b    = (const float*)d_in[2];
  const float* qkv_w  = (const float*)d_in[3];
  const float* qkv_b  = (const float*)d_in[4];
  const float* rpb    = (const float*)d_in[5];
  const float* proj_w = (const float*)d_in[6];
  const float* proj_b = (const float*)d_in[7];
  const float* c1w    = (const float*)d_in[8];
  const float* c1b    = (const float*)d_in[9];
  const float* c2w    = (const float*)d_in[10];
  const float* c2b    = (const float*)d_in[11];
  const float* ca1w   = (const float*)d_in[12];
  const float* ca1b   = (const float*)d_in[13];
  const float* ca2w   = (const float*)d_in[14];
  const float* ca2b   = (const float*)d_in[15];
  const float* n2g    = (const float*)d_in[16];
  const float* n2b    = (const float*)d_in[17];
  const float* fc1w   = (const float*)d_in[18];
  const float* fc1b   = (const float*)d_in[19];
  const float* fc2w   = (const float*)d_in[20];
  const float* fc2b   = (const float*)d_in[21];
  const int*   rpi    = (const int*)d_in[22];
  float* out = (float*)d_out;

  char* base = (char*)d_ws;
  size_t off = 0;
  auto take = [&](size_t bytes) -> char* {
    char* p = base + off;
    off = (off + bytes + 255) & ~(size_t)255;
    return p;
  };
  u16*   xn     = (u16*)take((size_t)L_TOK * 192 * 2);
  u16*   qkv    = (u16*)take((size_t)L_TOK * 576 * 2);
  u16*   vt     = (u16*)take((size_t)64 * 6 * 32 * 512 * 2);
  float* bb     = (float*)take((size_t)6 * 512 * 512 * 4);
  u16*   att    = (u16*)take((size_t)L_TOK * 192 * 2);
  float* x2     = (float*)take((size_t)L_TOK * 192 * 4);
  u16*   xn2    = (u16*)take((size_t)L_TOK * 192 * 2);
  u16*   mlph   = (u16*)take((size_t)L_TOK * 768 * 2);
  u16*   h1     = (u16*)take((size_t)L_TOK * 64 * 2);
  float* h2     = (float*)take((size_t)L_TOK * 192 * 4);
  float* pooled = (float*)take(192 * 4);
  float* avec   = (float*)take(192 * 4);
  u16*   Wqkv   = (u16*)take((size_t)192 * 576 * 2);
  u16*   Wproj  = (u16*)take((size_t)192 * 192 * 2);
  u16*   Wfc1   = (u16*)take((size_t)192 * 768 * 2);
  u16*   Wfc2   = (u16*)take((size_t)768 * 192 * 2);
  u16*   Wc1    = (u16*)take((size_t)27 * 192 * 64 * 2);
  u16*   Wc2    = (u16*)take((size_t)27 * 64 * 192 * 2);

  // Weight packing (bf16 B fragments)
  pack_b<<<432,  256, 0, stream>>>(qkv_w,  Wqkv,  192, 576, 1,  110592L);
  pack_b<<<144,  256, 0, stream>>>(proj_w, Wproj, 192, 192, 1,  36864L);
  pack_b<<<576,  256, 0, stream>>>(fc1w,   Wfc1,  192, 768, 1,  147456L);
  pack_b<<<576,  256, 0, stream>>>(fc2w,   Wfc2,  768, 192, 1,  147456L);
  pack_b<<<1296, 256, 0, stream>>>(c1w,    Wc1,   192, 64,  27, 331776L);
  pack_b<<<1296, 256, 0, stream>>>(c2w,    Wc2,   64,  192, 27, 331776L);

  // LN1 -> bf16 activations
  ln_k<<<4096, 256, 0, stream>>>(x, n1g, n1b, xn);

  // QKV GEMM (q pre-scaled by 1/sqrt(32)): M=32768, N=576 -> NT4=9
  gemm_k<0, 6><<<2304, 256, 0, stream>>>(xn, Wqkv, qkv_b, (void*)qkv, nullptr, 9, 192, 576);

  // V transpose + relative-position-bias gather
  vprep_k<<<24576, 256, 0, stream>>>(qkv, vt);
  bprep_k<<<6144,  256, 0, stream>>>(rpb, rpi, bb);

  // Window attention (flash-style)
  attn_k<<<1536, 256, 0, stream>>>(qkv, vt, bb, att);

  // proj GEMM -> x2 (f32): N=192 -> NT4=3
  gemm_k<1, 6><<<768, 256, 0, stream>>>(att, Wproj, proj_b, (void*)x2, nullptr, 3, 192, 192);

  // Conv branch: implicit GEMMs over 27 taps
  conv_k<0, 6><<<256, 256, 0, stream>>>(xn, Wc1, c1b, (void*)h1, 1, 192, 64);
  conv_k<1, 2><<<768, 256, 0, stream>>>(h1, Wc2, c2b, (void*)h2, 3, 64, 192);
  pool_k<<<192, 256, 0, stream>>>(h2, pooled);
  ca_k<<<1, 192, 0, stream>>>(pooled, ca1w, ca1b, ca2w, ca2b, avec);

  // Residual fuse + LN2
  fuse_k<<<4096, 256, 0, stream>>>(x, x2, h2, avec, n2g, n2b, xn2);

  // MLP: fc1 N=768 -> NT4=12 ; fc2 N=192 -> NT4=3, K=768 -> KT=24
  gemm_k<2, 6><<<3072, 256, 0, stream>>>(xn2, Wfc1, fc1b, (void*)mlph, nullptr, 12, 192, 768);
  gemm_k<3, 24><<<768, 256, 0, stream>>>(mlph, Wfc2, fc2b, (void*)out, x2, 3, 768, 192);
}
